// CosformerAttention_89386859364902
// MI455X (gfx1250) — matrix operand order repaired and checked by reading, not compiled, not run on hardware
//
#include <hip/hip_runtime.h>
#include <hip/hip_bf16.h>
#include <cmath>

typedef __attribute__((ext_vector_type(16))) _Float16 v16h;
typedef __attribute__((ext_vector_type(8)))  float    v8f;

#define HEADS 16
#define SEQ   1024
#define DIM   64
#define FEAT  128             // 2*DIM
#define CHUNK 64
#define NCH   (SEQ / CHUNK)   // 16
#define CP    (CHUNK + 8)     // padded rows: 144B, 16B aligned
#define FP    (FEAT + 8)      // padded rows: 272B, 16B aligned
#define DP    (DIM + 8)
#define DSP   (DIM + 4)       // f32 staging rows: 272B, 16B aligned

// 16x32 f16 A-fragment from row-major matrix. lane&15 -> M, lane>>4 -> K-half,
// VGPR j packs K pairs; contiguous 8-half runs -> two ds_load_b128.
__device__ inline v16h load_a_frag(const _Float16* mat, int stride, int m0, int k0, int lane) {
  const int half = lane >> 4;
  const _Float16* row = mat + (m0 + (lane & 15)) * stride;
  v16h a;
#pragma unroll
  for (int j = 0; j < 8; ++j) {
    const int kb = k0 + (j < 4 ? 2 * j : 16 + 2 * (j - 4)) + 8 * half;
    a[2 * j]     = row[kb];
    a[2 * j + 1] = row[kb + 1];
  }
  return a;
}

// 32x16 f16 B-fragment from row-major KxN matrix. lane&15 -> N column, lane>>4 -> K-half, VGPR j packs K pair.
__device__ inline v16h load_b_frag(const _Float16* mat, int stride, int k0, int n0, int lane) {
  const int half = lane >> 4;
  const _Float16* col = mat + k0 * stride + n0 + (lane & 15);
  v16h b;
#pragma unroll
  for (int j = 0; j < 8; ++j) {
    const int kb = (j < 4 ? 2 * j : 16 + 2 * (j - 4)) + 8 * half;
    b[2 * j]     = col[kb * stride];
    b[2 * j + 1] = col[(kb + 1) * stride];
  }
  return b;
}

// -------- Phase 1: per-(head,chunk) partial state S_c = K_^T @ V and ksum_c --------
__global__ __launch_bounds__(256) void cosformer_phase1(
    const float* __restrict__ kg, const float* __restrict__ vg,
    float* __restrict__ Spart, float* __restrict__ ksumws) {
  __shared__ _Float16 k_shT[FEAT][CP];    // 18.0 KB feature-major K_
  __shared__ _Float16 v_sh[CHUNK][DP];    //  9.0 KB
  __shared__ float    stage[FEAT][DSP];   // 34.0 KB f32 S staging

  const int blk = blockIdx.x;
  const int c = blk % NCH;
  const int tid = threadIdx.x;
  const size_t base = (size_t)blk * (CHUNK * DIM);
  const float4* kg4 = (const float4*)(kg + base);
  const float4* vg4 = (const float4*)(vg + base);

  for (int i4 = tid; i4 < CHUNK * DIM / 4; i4 += 256) {
    const int t = i4 >> 4, d = (i4 & 15) * 4;
    const int gpos = c * CHUNK + t;
    const float ang = 1.5707963267948966f * (float)(gpos + 1) * (1.0f / (float)SEQ);
    const float sn = __sinf(ang), cs = __cosf(ang);
    float4 kv = kg4[i4];
    kv.x = fmaxf(kv.x, 0.f); kv.y = fmaxf(kv.y, 0.f);
    kv.z = fmaxf(kv.z, 0.f); kv.w = fmaxf(kv.w, 0.f);
    k_shT[d + 0][t] = (_Float16)(kv.x * sn); k_shT[d + 0 + DIM][t] = (_Float16)(kv.x * cs);
    k_shT[d + 1][t] = (_Float16)(kv.y * sn); k_shT[d + 1 + DIM][t] = (_Float16)(kv.y * cs);
    k_shT[d + 2][t] = (_Float16)(kv.z * sn); k_shT[d + 2 + DIM][t] = (_Float16)(kv.z * cs);
    k_shT[d + 3][t] = (_Float16)(kv.w * sn); k_shT[d + 3 + DIM][t] = (_Float16)(kv.w * cs);
    const float4 vv = vg4[i4];
    v_sh[t][d + 0] = (_Float16)vv.x; v_sh[t][d + 1] = (_Float16)vv.y;
    v_sh[t][d + 2] = (_Float16)vv.z; v_sh[t][d + 3] = (_Float16)vv.w;
  }
  __syncthreads();

  const int wave = tid >> 5, lane = tid & 31, half = lane >> 4;
  const int mt = wave;                       // feature tile 0..7
  v8f acc[4] = {};

  for (int kk = 0; kk < CHUNK; kk += 32) {
    const v16h a = load_a_frag(&k_shT[0][0], CP, mt * 16, kk, lane);  // A = K_^T
#pragma unroll
    for (int nt = 0; nt < 4; ++nt) {
      const v16h b = load_b_frag(&v_sh[0][0], DP, kk, nt * 16, lane); // B = V
      acc[nt] = __builtin_amdgcn_wmma_f32_16x16x32_f16(
          false, a, false, b, (short)0, acc[nt], false, false);
    }
  }

  if (tid < FEAT) {                          // ksum while matmul lanes independent
    float s = 0.0f;
    for (int t = 0; t < CHUNK; ++t) s += (float)k_shT[tid][t];
    ksumws[(size_t)blk * FEAT + tid] = s;
  }

#pragma unroll
  for (int nt = 0; nt < 4; ++nt)
#pragma unroll
    for (int r = 0; r < 8; ++r)
      stage[mt * 16 + r + 8 * half][nt * 16 + (lane & 15)] = acc[nt][r];
  __syncthreads();

  float4* Sout4 = (float4*)(Spart + (size_t)blk * FEAT * DIM);
  for (int idx = tid; idx < FEAT * DIM / 4; idx += 256) {
    const int m = idx >> 4, dq = (idx & 15) * 4;
    Sout4[idx] = *(const float4*)&stage[m][dq];
  }
}

// -------- Phase 2: exclusive prefix over chunks (per head); prefix written as f16 --------
__global__ __launch_bounds__(256) void cosformer_phase2(
    const float* __restrict__ Spart, _Float16* __restrict__ Spref,
    float* __restrict__ ksumws) {
  const int h = blockIdx.x, tid = threadIdx.x;
  for (int e4 = tid; e4 < FEAT * DIM / 4; e4 += 256) {
    float ax = 0.f, ay = 0.f, az = 0.f, aw = 0.f;
    for (int c = 0; c < NCH; ++c) {
      const size_t off = (size_t)(h * NCH + c) * FEAT * DIM;
      const float4 t = ((const float4*)(Spart + off))[e4];
      _Float16* dst = Spref + off + (size_t)e4 * 4;
      dst[0] = (_Float16)ax; dst[1] = (_Float16)ay;
      dst[2] = (_Float16)az; dst[3] = (_Float16)aw;
      ax += t.x; ay += t.y; az += t.z; aw += t.w;
    }
  }
  if (tid < FEAT) {                          // in-place f32 prefix for ksum
    float acc = 0.0f;
    for (int c = 0; c < NCH; ++c) {
      float* p = ksumws + ((size_t)(h * NCH + c)) * FEAT + tid;
      const float t = *p; *p = acc; acc += t;
    }
  }
}

// -------- Phase 3: out = mask(Q_ K_^T) V + Q_ S_prefix, divided by denom --------
__global__ __launch_bounds__(256) void cosformer_phase3(
    const float* __restrict__ qg, const float* __restrict__ kg,
    const float* __restrict__ vg, const _Float16* __restrict__ Spref,
    const float* __restrict__ ksumws, float* __restrict__ outg) {
  __shared__ _Float16 q_sh[CHUNK][FP];    // 17.0 KB
  __shared__ _Float16 k_shT[FEAT][CP];    // 18.0 KB; reused: score tile, then f32 out stage
  __shared__ _Float16 v_sh[CHUNK][DP];    //  9.0 KB
  __shared__ _Float16 s_sh[FEAT][DIM];    // 16.0 KB
  __shared__ float ksum_sh[FEAT];
  __shared__ float dpart[CHUNK][4];
  __shared__ float denom_sh[CHUNK];

  const int blk = blockIdx.x;
  const int c = blk % NCH;
  const int tid = threadIdx.x;
  const size_t base = (size_t)blk * (CHUNK * DIM);
  const float4* qg4 = (const float4*)(qg + base);
  const float4* kg4 = (const float4*)(kg + base);
  const float4* vg4 = (const float4*)(vg + base);

  for (int i4 = tid; i4 < CHUNK * DIM / 4; i4 += 256) {
    const int t = i4 >> 4, d = (i4 & 15) * 4;
    const int gpos = c * CHUNK + t;
    const float ang = 1.5707963267948966f * (float)(gpos + 1) * (1.0f / (float)SEQ);
    const float sn = __sinf(ang), cs = __cosf(ang);
    float4 qv = qg4[i4];
    qv.x = fmaxf(qv.x, 0.f); qv.y = fmaxf(qv.y, 0.f);
    qv.z = fmaxf(qv.z, 0.f); qv.w = fmaxf(qv.w, 0.f);
    q_sh[t][d + 0] = (_Float16)(qv.x * sn); q_sh[t][d + 0 + DIM] = (_Float16)(qv.x * cs);
    q_sh[t][d + 1] = (_Float16)(qv.y * sn); q_sh[t][d + 1 + DIM] = (_Float16)(qv.y * cs);
    q_sh[t][d + 2] = (_Float16)(qv.z * sn); q_sh[t][d + 2 + DIM] = (_Float16)(qv.z * cs);
    q_sh[t][d + 3] = (_Float16)(qv.w * sn); q_sh[t][d + 3 + DIM] = (_Float16)(qv.w * cs);
    float4 kv = kg4[i4];
    kv.x = fmaxf(kv.x, 0.f); kv.y = fmaxf(kv.y, 0.f);
    kv.z = fmaxf(kv.z, 0.f); kv.w = fmaxf(kv.w, 0.f);
    k_shT[d + 0][t] = (_Float16)(kv.x * sn); k_shT[d + 0 + DIM][t] = (_Float16)(kv.x * cs);
    k_shT[d + 1][t] = (_Float16)(kv.y * sn); k_shT[d + 1 + DIM][t] = (_Float16)(kv.y * cs);
    k_shT[d + 2][t] = (_Float16)(kv.z * sn); k_shT[d + 2 + DIM][t] = (_Float16)(kv.z * cs);
    k_shT[d + 3][t] = (_Float16)(kv.w * sn); k_shT[d + 3 + DIM][t] = (_Float16)(kv.w * cs);
    const float4 vv = vg4[i4];
    v_sh[t][d + 0] = (_Float16)vv.x; v_sh[t][d + 1] = (_Float16)vv.y;
    v_sh[t][d + 2] = (_Float16)vv.z; v_sh[t][d + 3] = (_Float16)vv.w;
  }
  {   // S prefix: already f16 — straight 16B copies into LDS
    const uint4* Sp4 = (const uint4*)(Spref + (size_t)blk * FEAT * DIM);
    uint4* s4 = (uint4*)&s_sh[0][0];
    for (int idx = tid; idx < FEAT * DIM / 8; idx += 256) s4[idx] = Sp4[idx];
  }
  if (tid < FEAT) ksum_sh[tid] = ksumws[(size_t)blk * FEAT + tid];
  __syncthreads();

  const int wave = tid >> 5, lane = tid & 31, half = lane >> 4;
  const int mt = wave & 3;               // query tile 0..3
  const int ntBase = (wave >> 2) * 2;    // two key/dim tiles per wave

  // ---- scores: Q_ @ K_^T (K = FEAT) ----
  v8f att[2] = {};
  for (int kk = 0; kk < FEAT; kk += 32) {
    const v16h a = load_a_frag(&q_sh[0][0], FP, mt * 16, kk, lane);
#pragma unroll
    for (int p = 0; p < 2; ++p) {
      const v16h b = load_b_frag(&k_shT[0][0], CP, kk, (ntBase + p) * 16, lane);
      att[p] = __builtin_amdgcn_wmma_f32_16x16x32_f16(
          false, a, false, b, (short)0, att[p], false, false);
    }
  }
  __syncthreads();   // done reading k_shT

  _Float16 (*sc_sh)[CP] = (_Float16(*)[CP])&k_shT[0][0];   // masked score tile alias
#pragma unroll
  for (int p = 0; p < 2; ++p) {
#pragma unroll
    for (int r = 0; r < 8; ++r) {
      const int tq = mt * 16 + r + 8 * half;
      const int tk = (ntBase + p) * 16 + (lane & 15);
      sc_sh[tq][tk] = (_Float16)((tk <= tq) ? att[p][r] : 0.0f);
    }
  }
  __syncthreads();

  // ---- denominator: rowsum(masked scores) + q_ . ksum_prefix (4 threads/row) ----
  {
    const int row = tid & 63, part = tid >> 6;
    float dsum = 0.0f;
    for (int tk = 16 * part; tk < 16 * part + 16; ++tk) dsum += (float)sc_sh[row][tk];
    for (int f = 32 * part; f < 32 * part + 32; ++f)
      dsum += (float)q_sh[row][f] * ksum_sh[f];
    dpart[row][part] = dsum;
  }
  __syncthreads();
  if (tid < CHUNK)
    denom_sh[tid] = fmaxf(dpart[tid][0] + dpart[tid][1] + dpart[tid][2] + dpart[tid][3], 1e-6f);
  __syncthreads();

  // ---- output: inter part first (q_sh, s_sh), then intra (sc_sh, v_sh) ----
  v8f oacc[2] = {};
  for (int kk = 0; kk < FEAT; kk += 32) {
    const v16h a = load_a_frag(&q_sh[0][0], FP, mt * 16, kk, lane);
#pragma unroll
    for (int p = 0; p < 2; ++p) {
      const v16h b = load_b_frag(&s_sh[0][0], DIM, kk, (ntBase + p) * 16, lane);
      oacc[p] = __builtin_amdgcn_wmma_f32_16x16x32_f16(
          false, a, false, b, (short)0, oacc[p], false, false);
    }
  }
  for (int kk = 0; kk < CHUNK; kk += 32) {
    const v16h a = load_a_frag(&sc_sh[0][0], CP, mt * 16, kk, lane);
#pragma unroll
    for (int p = 0; p < 2; ++p) {
      const v16h b = load_b_frag(&v_sh[0][0], DP, kk, (ntBase + p) * 16, lane);
      oacc[p] = __builtin_amdgcn_wmma_f32_16x16x32_f16(
          false, a, false, b, (short)0, oacc[p], false, false);
    }
  }
  __syncthreads();   // all waves done with sc_sh; reuse region as f32 out stage

  float (*ostage)[DSP] = (float(*)[DSP])&k_shT[0][0];   // 64*68*4 = 17 KB <= 18 KB
#pragma unroll
  for (int p = 0; p < 2; ++p) {
#pragma unroll
    for (int r = 0; r < 8; ++r) {
      const int t = mt * 16 + r + 8 * half;
      const int d = (ntBase + p) * 16 + (lane & 15);
      ostage[t][d] = oacc[p][r] / denom_sh[t];
    }
  }
  __syncthreads();

  float4* out4 = (float4*)(outg + base);
  for (int idx = tid; idx < CHUNK * DIM / 4; idx += 256) {
    const int t = idx >> 4, dq = (idx & 15) * 4;
    out4[idx] = *(const float4*)&ostage[t][dq];
  }
}

extern "C" void kernel_launch(void* const* d_in, const int* in_sizes, int n_in,
                              void* d_out, int out_size, void* d_ws, size_t ws_size,
                              hipStream_t stream) {
  const float* q = (const float*)d_in[0];
  const float* k = (const float*)d_in[1];
  const float* v = (const float*)d_in[2];
  float* out = (float*)d_out;

  // workspace: f32 S partials (8 MB) | f16 S prefixes (4 MB) | f32 ksum (128 KB)
  float*     Spart  = (float*)d_ws;
  _Float16*  Spref  = (_Float16*)(Spart + (size_t)HEADS * NCH * FEAT * DIM);
  float*     ksumws = (float*)(Spref + (size_t)HEADS * NCH * FEAT * DIM);

  cosformer_phase1<<<HEADS * NCH, 256, 0, stream>>>(k, v, Spart, ksumws);
  cosformer_phase2<<<HEADS, 256, 0, stream>>>(Spart, Spref, ksumws);
  cosformer_phase3<<<HEADS * NCH, 256, 0, stream>>>(q, k, v, Spref, ksumws, out);
}
